// NS_transformer_84859963835059
// MI455X (gfx1250) — matrix operand order repaired and checked by reading, not compiled, not run on hardware
//
#include <hip/hip_runtime.h>
#include <math.h>

// ---------------------------------------------------------------------------
// NS-Transformer forward for MI455X (gfx1250, wave32, WMMA).
// Compute-bound (~350 GFLOP vs ~50MB hot working set @23.3TB/s) =>
// all big GEMMs + attention go through v_wmma_f32_16x16x32_f16.
// Attention is fused (flash-style) so the 268MB score tensor never exists.
// GEMM uses a 32x64 wave tile: 8 WMMAs per 32-wide K step, B tile staged
// once in LDS per block and reused by 2 A fragments per wave.
// ---------------------------------------------------------------------------

#define D_MODEL 512
#define N_HEADS 8
#define HEAD_DIM 64
#define D_FF 2048
#define BATCH 32
#define SEQ 512
#define CIN 21
#define PRED 256
#define MROWS (BATCH * SEQ)   // 16384

typedef __attribute__((ext_vector_type(16))) _Float16 v16h;
typedef __attribute__((ext_vector_type(8)))  float    v8f;

__device__ __forceinline__ v8f wmma_f16(v16h a, v16h b, v8f c) {
    return __builtin_amdgcn_wmma_f32_16x16x32_f16(false, a, false, b, (short)0, c, false, false);
}
__device__ __forceinline__ v8f v8f_zero() {
    v8f z; for (int i = 0; i < 8; ++i) z[i] = 0.f; return z;
}
__device__ __forceinline__ float gelu_exact(float x) {
    return 0.5f * x * (1.0f + erff(x * 0.70710678118f));
}

// ---------------------------------------------------------------------------
// GEMM: Y[M,N] = epi(X[M,K] @ W[K,N] + bias [, + Res]) ; f32 in, f16 WMMA, f32 acc.
// Block = 256 thr (8 waves). Block tile 256(M) x 64(N). Wave tile 32 x 64.
// W tile (32x64) staged to LDS as f16 (shared by all 8 waves).
// EPI: 0 = bias, 1 = bias+GELU(exact), 2 = bias+residual
// ---------------------------------------------------------------------------
template <int EPI>
__global__ __launch_bounds__(256) void gemm_wmma(
    const float* __restrict__ X, const float* __restrict__ W,
    const float* __restrict__ bias, const float* __restrict__ Res,
    float* __restrict__ Y, int M, int N, int K)
{
    __shared__ _Float16 Bs[32][72];      // 32(K) x 64(N), padded stride

    const int tid  = threadIdx.x;
    const int lane = tid & 31;
    const int wave = tid >> 5;
    const int half = (lane >> 4) & 1;
    const int l15  = lane & 15;

    const int m0 = blockIdx.x * 256 + wave * 32;
    const int n0 = blockIdx.y * 64;

    v8f acc[2][4];
    for (int mi = 0; mi < 2; ++mi)
        for (int nc = 0; nc < 4; ++nc) acc[mi][nc] = v8f_zero();

    for (int k0 = 0; k0 < K; k0 += 32) {
        // cooperative stage of W tile: 2048 f32 -> f16, 8 per thread, coalesced
        {
            const int base = tid * 8;
            const int kk = base >> 6;
            const int nn = base & 63;
            const float* wp = W + (size_t)(k0 + kk) * N + n0 + nn;
            #pragma unroll
            for (int j = 0; j < 8; ++j) Bs[kk][nn + j] = (_Float16)wp[j];
        }
        __syncthreads();

        // A fragments: 16x32 f16 each (lane<16 holds K {0-15},
        // lane>=16 holds K {16-31}, row = lane&15)
        v16h a[2];
        #pragma unroll
        for (int mi = 0; mi < 2; ++mi) {
            const float* xrow = X + (size_t)(m0 + mi * 16 + l15) * K + k0;
            #pragma unroll
            for (int e = 0; e < 16; ++e)
                a[mi][e] = (_Float16)xrow[half * 16 + e];
        }

        #pragma unroll
        for (int nc = 0; nc < 4; ++nc) {
            v16h bf;
            #pragma unroll
            for (int e = 0; e < 16; ++e)
                bf[e] = Bs[half * 16 + e][nc * 16 + l15];
            acc[0][nc] = wmma_f16(a[0], bf, acc[0][nc]);
            acc[1][nc] = wmma_f16(a[1], bf, acc[1][nc]);
        }
        __syncthreads();
    }

    // epilogue (C layout: vgpr r -> row r + half*8, col = lane&15 within chunk)
    for (int mi = 0; mi < 2; ++mi)
        for (int nc = 0; nc < 4; ++nc) {
            const int col = n0 + nc * 16 + l15;
            const float bv = bias[col];
            for (int r = 0; r < 8; ++r) {
                const int row = m0 + mi * 16 + r + half * 8;
                float v = acc[mi][nc][r] + bv;
                if (EPI == 1) v = gelu_exact(v);
                if (EPI == 2) v += Res[(size_t)row * N + col];
                Y[(size_t)row * N + col] = v;
            }
        }
}

// ---------------------------------------------------------------------------
// Fused DS-attention (flash style), one wave per 16 query rows per (b,h).
// scores = scale*(Q Kᵀ * tau[b] + delta[b,s]) ; optional causal; online softmax;
// O = softmax @ V.  Q/K/V/O layout: [B*L, D_MODEL] with head h at cols h*64..
// ---------------------------------------------------------------------------
__global__ __launch_bounds__(32) void attn_fused(
    const float* __restrict__ Q, const float* __restrict__ K,
    const float* __restrict__ V, float* __restrict__ O,
    const float* __restrict__ tau, const float* __restrict__ delta,
    int causal, int Lq, int Lk)
{
    const int bh = blockIdx.y;
    const int b  = bh / N_HEADS;
    const int h  = bh % N_HEADS;
    const int l0 = blockIdx.x * 16;

    const int lane = threadIdx.x;
    const int half = (lane >> 4) & 1;
    const int l15  = lane & 15;

    const float taub  = tau[b];
    const float scale = 0.125f;   // 1/sqrt(64)

    __shared__ float    s_tile[16][33];
    __shared__ _Float16 p_tile[16][32];
    __shared__ float    row_f[16];
    __shared__ float    row_inv[16];

    // preload Q fragments for both 32-wide K-dim chunks (HEAD_DIM = 64)
    v16h qa[2];
    {
        const float* qrow = Q + ((size_t)(b * Lq + l0 + l15) * D_MODEL) + h * HEAD_DIM;
        for (int kc = 0; kc < 2; ++kc)
            for (int e = 0; e < 16; ++e)
                qa[kc][e] = (_Float16)qrow[kc * 32 + half * 16 + e];
    }

    v8f o_acc[4];
    for (int i = 0; i < 4; ++i) o_acc[i] = v8f_zero();
    float m_run = -__builtin_inff();
    float l_run = 0.f;

    for (int s0 = 0; s0 < Lk; s0 += 32) {
        if (causal && s0 > l0 + 15) break;

        // ---- score tile 16(l) x 32(s) via 4 WMMAs
        for (int nc = 0; nc < 2; ++nc) {
            v8f sc = v8f_zero();
            const int scol = s0 + nc * 16 + l15;
            const float* krow = K + ((size_t)(b * Lk + scol) * D_MODEL) + h * HEAD_DIM;
            for (int kc = 0; kc < 2; ++kc) {
                v16h kf;
                for (int e = 0; e < 16; ++e)
                    kf[e] = (_Float16)krow[kc * 32 + half * 16 + e];
                sc = wmma_f16(qa[kc], kf, sc);
            }
            for (int r = 0; r < 8; ++r)
                s_tile[r + half * 8][nc * 16 + l15] = sc[r];
        }
        __syncthreads();

        // ---- online softmax update: lane owns row l15, cols [half*16, half*16+16)
        {
            const int row = l15;
            const int lg  = l0 + row;
            float vals[16];
            float mx = -__builtin_inff();
            for (int j = 0; j < 16; ++j) {
                const int s = s0 + half * 16 + j;
                float v = s_tile[row][half * 16 + j] * taub;
                if (delta) v += delta[b * Lk + s];
                v *= scale;
                if (causal && s > lg) v = -__builtin_inff();
                vals[j] = v;
                mx = fmaxf(mx, v);
            }
            mx = fmaxf(mx, __shfl_xor(mx, 16, 32));
            const float m_new = fmaxf(m_run, mx);
            float sum = 0.f;
            for (int j = 0; j < 16; ++j) {
                const float p = __expf(vals[j] - m_new);
                sum += p;
                p_tile[row][half * 16 + j] = (_Float16)p;
            }
            sum += __shfl_xor(sum, 16, 32);
            const float f = __expf(m_run - m_new);
            l_run = l_run * f + sum;
            m_run = m_new;
            if (half == 0) row_f[row] = f;
        }
        __syncthreads();

        // ---- rescale accumulators, then P(16x32) @ V(32x64) via 4 WMMAs
        for (int dc = 0; dc < 4; ++dc)
            for (int r = 0; r < 8; ++r)
                o_acc[dc][r] *= row_f[r + half * 8];

        v16h pf;
        for (int e = 0; e < 16; ++e)
            pf[e] = p_tile[l15][half * 16 + e];
        for (int dc = 0; dc < 4; ++dc) {
            v16h vf;
            for (int e = 0; e < 16; ++e) {
                const int s = s0 + half * 16 + e;
                vf[e] = (_Float16)V[((size_t)(b * Lk + s) * D_MODEL) + h * HEAD_DIM + dc * 16 + l15];
            }
            o_acc[dc] = wmma_f16(pf, vf, o_acc[dc]);
        }
        __syncthreads();
    }

    if (half == 0) row_inv[l15] = 1.f / l_run;
    __syncthreads();
    for (int dc = 0; dc < 4; ++dc)
        for (int r = 0; r < 8; ++r) {
            const int row = r + half * 8;
            O[((size_t)(b * Lq + l0 + row) * D_MODEL) + h * HEAD_DIM + dc * 16 + l15]
                = o_acc[dc][r] * row_inv[row];
        }
}

// ---------------------------------------------------------------------------
// LayerNorm over last dim (D=512), one row per block.
// ---------------------------------------------------------------------------
__global__ __launch_bounds__(256) void layernorm_k(
    const float* __restrict__ X, const float* __restrict__ g,
    const float* __restrict__ bta, float* __restrict__ Y, int D)
{
    const int row = blockIdx.x;
    const float* x = X + (size_t)row * D;
    __shared__ float red[256];

    float s = 0.f;
    for (int i = threadIdx.x; i < D; i += 256) s += x[i];
    red[threadIdx.x] = s; __syncthreads();
    for (int st = 128; st > 0; st >>= 1) {
        if (threadIdx.x < st) red[threadIdx.x] += red[threadIdx.x + st];
        __syncthreads();
    }
    const float mean = red[0] / D;
    __syncthreads();

    float s2 = 0.f;
    for (int i = threadIdx.x; i < D; i += 256) { float d = x[i] - mean; s2 += d * d; }
    red[threadIdx.x] = s2; __syncthreads();
    for (int st = 128; st > 0; st >>= 1) {
        if (threadIdx.x < st) red[threadIdx.x] += red[threadIdx.x + st];
        __syncthreads();
    }
    const float rstd = rsqrtf(red[0] / D + 1e-5f);
    __syncthreads();

    for (int i = threadIdx.x; i < D; i += 256)
        Y[(size_t)row * D + i] = (x[i] - mean) * rstd * g[i] + bta[i];
}

// ---------------------------------------------------------------------------
// Per-(b,c) mean/std of x_enc over time axis.
// ---------------------------------------------------------------------------
__global__ __launch_bounds__(256) void enc_stats(
    const float* __restrict__ x, float* __restrict__ meanv, float* __restrict__ stdv)
{
    const int b = blockIdx.x / CIN, c = blockIdx.x % CIN;
    __shared__ float r1[256], r2[256];
    float s = 0.f, s2 = 0.f;
    for (int t = threadIdx.x; t < SEQ; t += 256) {
        const float v = x[((size_t)b * SEQ + t) * CIN + c];
        s += v; s2 += v * v;
    }
    r1[threadIdx.x] = s; r2[threadIdx.x] = s2; __syncthreads();
    for (int st = 128; st > 0; st >>= 1) {
        if (threadIdx.x < st) { r1[threadIdx.x] += r1[threadIdx.x + st]; r2[threadIdx.x] += r2[threadIdx.x + st]; }
        __syncthreads();
    }
    if (threadIdx.x == 0) {
        const float m = r1[0] / SEQ;
        const float var = r2[0] / SEQ - m * m;
        meanv[b * CIN + c] = m;
        stdv[b * CIN + c]  = sqrtf(var + 1e-5f);
    }
}

__global__ void normalize_k(const float* __restrict__ x, const float* __restrict__ meanv,
                            const float* __restrict__ stdv, float* __restrict__ y)
{
    const int idx = blockIdx.x * 256 + threadIdx.x;
    if (idx >= BATCH * SEQ * CIN) return;
    const int c = idx % CIN;
    const int b = idx / (SEQ * CIN);
    y[idx] = (x[idx] - meanv[b * CIN + c]) / stdv[b * CIN + c];
}

// ---------------------------------------------------------------------------
// Projector series_conv: conv over feature axis (circular), channels = SEQ.
// out[b,e] = sum_{k,s} x[b,s,wrap(e-1+k)] * w[k*SEQ + s]
// ---------------------------------------------------------------------------
__global__ __launch_bounds__(256) void series_conv(
    const float* __restrict__ x, const float* __restrict__ w, float* __restrict__ out)
{
    const int b = blockIdx.x / CIN, e = blockIdx.x % CIN;
    __shared__ float red[256];
    float acc = 0.f;
    for (int idx = threadIdx.x; idx < 3 * SEQ; idx += 256) {
        const int k = idx / SEQ, s = idx % SEQ;
        const int f = (e - 1 + k + CIN) % CIN;
        acc += x[((size_t)b * SEQ + s) * CIN + f] * w[k * SEQ + s];
    }
    red[threadIdx.x] = acc; __syncthreads();
    for (int st = 128; st > 0; st >>= 1) {
        if (threadIdx.x < st) red[threadIdx.x] += red[threadIdx.x + st];
        __syncthreads();
    }
    if (threadIdx.x == 0) out[b * CIN + e] = red[0];
}

// Projector MLP: [conv_out, stats] (42) -> 128 relu -> 128 relu -> out_dim.
__global__ __launch_bounds__(128) void proj_mlp(
    const float* __restrict__ conv_out, const float* __restrict__ stats,
    const float* __restrict__ w0, const float* __restrict__ b0,
    const float* __restrict__ w1, const float* __restrict__ b1,
    const float* __restrict__ w2, float* __restrict__ out,
    int out_dim, int apply_exp)
{
    const int b = blockIdx.x;
    __shared__ float in[48], h0[128], h1[128];
    const int tid = threadIdx.x;
    if (tid < CIN)           in[tid] = conv_out[b * CIN + tid];
    else if (tid < 2 * CIN)  in[tid] = stats[b * CIN + tid - CIN];
    __syncthreads();
    {
        float a = b0[tid];
        for (int i = 0; i < 2 * CIN; ++i) a += in[i] * w0[i * 128 + tid];
        h0[tid] = fmaxf(a, 0.f);
    }
    __syncthreads();
    {
        float a = b1[tid];
        for (int i = 0; i < 128; ++i) a += h0[i] * w1[i * 128 + tid];
        h1[tid] = fmaxf(a, 0.f);
    }
    __syncthreads();
    for (int o = tid; o < out_dim; o += 128) {
        float a = 0.f;
        for (int i = 0; i < 128; ++i) a += h1[i] * w2[i * out_dim + o];
        out[b * out_dim + o] = apply_exp ? __expf(a) : a;
    }
}

// ---------------------------------------------------------------------------
// Token embedding: circular conv1d (K=3) over time + sinusoidal PE (computed).
// y[b,t,d] = sum_{k,c} x[b,wrap(t-1+k),c]*w[(k*CIN+c)*512 + d] + PE(t,d)
// ---------------------------------------------------------------------------
__global__ __launch_bounds__(256) void embed_conv_pe(
    const float* __restrict__ x, const float* __restrict__ w, float* __restrict__ y)
{
    const size_t idx = (size_t)blockIdx.x * 256 + threadIdx.x;   // B*S*512
    const int d = (int)(idx % D_MODEL);
    const size_t bt = idx / D_MODEL;
    const int t = (int)(bt % SEQ);
    const int b = (int)(bt / SEQ);

    float acc = 0.f;
    for (int k = 0; k < 3; ++k) {
        const int tt = (t - 1 + k + SEQ) % SEQ;
        const float* xr = x + ((size_t)b * SEQ + tt) * CIN;
        const float* wr = w + (size_t)k * CIN * D_MODEL + d;
        for (int c = 0; c < CIN; ++c) acc += xr[c] * wr[c * D_MODEL];
    }
    const float div = __expf(-(float)(d & ~1) * (9.210340372f / 512.f));
    const float ang = (float)t * div;
    acc += (d & 1) ? __cosf(ang) : __sinf(ang);
    y[idx] = acc;
}

// ---------------------------------------------------------------------------
// Final: out = (x @ out_w + out_b) * std + mean, slice last PRED steps.
// ---------------------------------------------------------------------------
__global__ __launch_bounds__(256) void out_proj(
    const float* __restrict__ x, const float* __restrict__ w, const float* __restrict__ bias,
    const float* __restrict__ stdv, const float* __restrict__ meanv, float* __restrict__ out)
{
    const int idx = blockIdx.x * 256 + threadIdx.x;
    if (idx >= BATCH * PRED * CIN) return;
    const int c  = idx % CIN;
    const int tp = (idx / CIN) % PRED;
    const int b  = idx / (CIN * PRED);
    const int t  = (SEQ - PRED) + tp;
    const float* xr = x + ((size_t)b * SEQ + t) * D_MODEL;
    float acc = bias[c];
    for (int d = 0; d < D_MODEL; ++d) acc += xr[d] * w[d * CIN + c];
    out[idx] = acc * stdv[b * CIN + c] + meanv[b * CIN + c];
}

// ---------------------------------------------------------------------------
// Parameter index map: harness flattens setup_inputs() -> x_enc, x_dec, then
// params as a JAX pytree (dict keys sorted alphabetically at every level).
// ---------------------------------------------------------------------------
enum {
    IN_XENC = 0, IN_XDEC = 1,
    P_DEC_EMB = 2,
    DL_B1 = 3, DL_B2 = 4,
    DL_CR_BK = 5, DL_CR_BO = 6, DL_CR_BQ = 7, DL_CR_BV = 8,
    DL_CR_WK = 9, DL_CR_WO = 10, DL_CR_WQ = 11, DL_CR_WV = 12,
    DL_LN1B = 13, DL_LN1G = 14, DL_LN2B = 15, DL_LN2G = 16, DL_LN3B = 17, DL_LN3G = 18,
    DL_SE_BK = 19, DL_SE_BO = 20, DL_SE_BQ = 21, DL_SE_BV = 22,
    DL_SE_WK = 23, DL_SE_WO = 24, DL_SE_WQ = 25, DL_SE_WV = 26,
    DL_W1 = 27, DL_W2 = 28,
    P_DEC_NB = 29, P_DEC_NG = 30,
    DELTA_B0 = 31, DELTA_B1 = 32, DELTA_CONV = 33, DELTA_W0 = 34, DELTA_W1 = 35, DELTA_W2 = 36,
    P_ENC_EMB = 37,
    EL0 = 38, EL1 = 54,   // per enc layer: bk,bo,bq,bv,wk,wo,wq,wv,b1,b2,ln1b,ln1g,ln2b,ln2g,w1,w2
    P_ENC_NB = 70, P_ENC_NG = 71,
    P_OUT_B = 72, P_OUT_W = 73,
    TAU_B0 = 74, TAU_B1 = 75, TAU_CONV = 76, TAU_W0 = 77, TAU_W1 = 78, TAU_W2 = 79,
};

extern "C" void kernel_launch(void* const* d_in, const int* in_sizes, int n_in,
                              void* d_out, int out_size, void* d_ws, size_t ws_size,
                              hipStream_t stream)
{
    (void)in_sizes; (void)n_in; (void)out_size; (void)ws_size;
    auto P = [&](int i) -> const float* { return (const float*)d_in[i]; };
    const float* x_enc = P(IN_XENC);
    const float* x_dec = P(IN_XDEC);

    // workspace carving (floats)
    float* ws = (float*)d_ws;
    size_t off = 0;
    auto alloc = [&](size_t n) { float* p = ws + off; off += n; return p; };
    const size_t SZ = (size_t)MROWS * D_MODEL;
    float* bx   = alloc(SZ);                      // running activation
    float* bq   = alloc(SZ);
    float* bk   = alloc(SZ);
    float* bv   = alloc(SZ);
    float* bt   = alloc(SZ);                      // attn out / ffn out
    float* bh   = alloc((size_t)MROWS * D_FF);    // ffn hidden
    float* benc = alloc(SZ);                      // encoder output (cross K/V)
    float* bdec = alloc(SZ);                      // decoder activation
    float* xe    = alloc((size_t)BATCH * SEQ * CIN);
    float* meanE = alloc(BATCH * CIN);
    float* stdE  = alloc(BATCH * CIN);
    float* tauv  = alloc(BATCH);
    float* deltav= alloc(BATCH * SEQ);
    float* convt = alloc(BATCH * CIN);

    const dim3 gGemm512(MROWS / 256, D_MODEL / 64);
    const dim3 gGemmFF1(MROWS / 256, D_FF / 64);
    const dim3 gAttn(SEQ / 16, BATCH * N_HEADS);

    // ---- stats + normalization ----
    enc_stats<<<BATCH * CIN, 256, 0, stream>>>(x_enc, meanE, stdE);
    normalize_k<<<(BATCH * SEQ * CIN + 255) / 256, 256, 0, stream>>>(x_enc, meanE, stdE, xe);

    // ---- tau / delta projectors (on raw x_enc) ----
    series_conv<<<BATCH * CIN, 256, 0, stream>>>(x_enc, P(TAU_CONV), convt);
    proj_mlp<<<BATCH, 128, 0, stream>>>(convt, stdE, P(TAU_W0), P(TAU_B0),
                                        P(TAU_W1), P(TAU_B1), P(TAU_W2), tauv, 1, 1);
    series_conv<<<BATCH * CIN, 256, 0, stream>>>(x_enc, P(DELTA_CONV), convt);
    proj_mlp<<<BATCH, 128, 0, stream>>>(convt, meanE, P(DELTA_W0), P(DELTA_B0),
                                        P(DELTA_W1), P(DELTA_B1), P(DELTA_W2), deltav, SEQ, 0);

    // ---- encoder embedding ----
    embed_conv_pe<<<(unsigned)(SZ / 256), 256, 0, stream>>>(xe, P(P_ENC_EMB), bx);

    // ---- encoder layers ----
    for (int L = 0; L < 2; ++L) {
        const int base = (L == 0) ? EL0 : EL1;
        gemm_wmma<0><<<gGemm512, 256, 0, stream>>>(bx, P(base + 6), P(base + 2), nullptr, bq, MROWS, D_MODEL, D_MODEL); // Q
        gemm_wmma<0><<<gGemm512, 256, 0, stream>>>(bx, P(base + 4), P(base + 0), nullptr, bk, MROWS, D_MODEL, D_MODEL); // K
        gemm_wmma<0><<<gGemm512, 256, 0, stream>>>(bx, P(base + 7), P(base + 3), nullptr, bv, MROWS, D_MODEL, D_MODEL); // V
        attn_fused<<<gAttn, 32, 0, stream>>>(bq, bk, bv, bt, tauv, deltav, /*causal=*/0, SEQ, SEQ);
        gemm_wmma<2><<<gGemm512, 256, 0, stream>>>(bt, P(base + 5), P(base + 1), bx, bq, MROWS, D_MODEL, D_MODEL);      // O + res
        layernorm_k<<<MROWS, 256, 0, stream>>>(bq, P(base + 11), P(base + 10), bx, D_MODEL);                            // ln1
        gemm_wmma<1><<<gGemmFF1, 256, 0, stream>>>(bx, P(base + 14), P(base + 8), nullptr, bh, MROWS, D_FF, D_MODEL);   // ffn1+gelu
        gemm_wmma<2><<<gGemm512, 256, 0, stream>>>(bh, P(base + 15), P(base + 9), bx, bt, MROWS, D_MODEL, D_FF);        // ffn2+res
        layernorm_k<<<MROWS, 256, 0, stream>>>(bt, P(base + 13), P(base + 12), bx, D_MODEL);                            // ln2
    }
    layernorm_k<<<MROWS, 256, 0, stream>>>(bx, P(P_ENC_NG), P(P_ENC_NB), benc, D_MODEL);

    // ---- decoder embedding ----
    embed_conv_pe<<<(unsigned)(SZ / 256), 256, 0, stream>>>(x_dec, P(P_DEC_EMB), bdec);

    // ---- decoder layer: causal self-attn (delta=None) ----
    gemm_wmma<0><<<gGemm512, 256, 0, stream>>>(bdec, P(DL_SE_WQ), P(DL_SE_BQ), nullptr, bq, MROWS, D_MODEL, D_MODEL);
    gemm_wmma<0><<<gGemm512, 256, 0, stream>>>(bdec, P(DL_SE_WK), P(DL_SE_BK), nullptr, bk, MROWS, D_MODEL, D_MODEL);
    gemm_wmma<0><<<gGemm512, 256, 0, stream>>>(bdec, P(DL_SE_WV), P(DL_SE_BV), nullptr, bv, MROWS, D_MODEL, D_MODEL);
    attn_fused<<<gAttn, 32, 0, stream>>>(bq, bk, bv, bt, tauv, nullptr, /*causal=*/1, SEQ, SEQ);
    gemm_wmma<2><<<gGemm512, 256, 0, stream>>>(bt, P(DL_SE_WO), P(DL_SE_BO), bdec, bq, MROWS, D_MODEL, D_MODEL);
    layernorm_k<<<MROWS, 256, 0, stream>>>(bq, P(DL_LN1G), P(DL_LN1B), bdec, D_MODEL);

    // ---- decoder cross-attn (K/V from encoder, delta applied) ----
    gemm_wmma<0><<<gGemm512, 256, 0, stream>>>(bdec, P(DL_CR_WQ), P(DL_CR_BQ), nullptr, bq, MROWS, D_MODEL, D_MODEL);
    gemm_wmma<0><<<gGemm512, 256, 0, stream>>>(benc, P(DL_CR_WK), P(DL_CR_BK), nullptr, bk, MROWS, D_MODEL, D_MODEL);
    gemm_wmma<0><<<gGemm512, 256, 0, stream>>>(benc, P(DL_CR_WV), P(DL_CR_BV), nullptr, bv, MROWS, D_MODEL, D_MODEL);
    attn_fused<<<gAttn, 32, 0, stream>>>(bq, bk, bv, bt, tauv, deltav, /*causal=*/0, SEQ, SEQ);
    gemm_wmma<2><<<gGemm512, 256, 0, stream>>>(bt, P(DL_CR_WO), P(DL_CR_BO), bdec, bq, MROWS, D_MODEL, D_MODEL);
    layernorm_k<<<MROWS, 256, 0, stream>>>(bq, P(DL_LN2G), P(DL_LN2B), bdec, D_MODEL);

    // ---- decoder FFN ----
    gemm_wmma<1><<<gGemmFF1, 256, 0, stream>>>(bdec, P(DL_W1), P(DL_B1), nullptr, bh, MROWS, D_FF, D_MODEL);
    gemm_wmma<2><<<gGemm512, 256, 0, stream>>>(bh, P(DL_W2), P(DL_B2), bdec, bt, MROWS, D_MODEL, D_FF);
    layernorm_k<<<MROWS, 256, 0, stream>>>(bt, P(DL_LN3G), P(DL_LN3B), bdec, D_MODEL);

    // ---- final norm + output projection + de-normalize + slice ----
    layernorm_k<<<MROWS, 256, 0, stream>>>(bdec, P(P_DEC_NG), P(P_DEC_NB), bt, D_MODEL);
    out_proj<<<(BATCH * PRED * CIN + 255) / 256, 256, 0, stream>>>(
        bt, P(P_OUT_W), P(P_OUT_B), stdE, meanE, (float*)d_out);
}
